// GenerativeRowRNN_82429012345537
// MI455X (gfx1250) — matrix operand order repaired and checked by reading, not compiled, not run on hardware
//
#include <hip/hip_runtime.h>
#include <hip/hip_bf16.h>
#include <stdint.h>

// ---------------------------------------------------------------------------
// GenerativeRowRNN on MI455X (gfx1250, wave32, WMMA bf16 16x16x32)
//
// B=128, C=3, R=CO=32 -> Nr=Nc=31, E=H=256, V=257.
//
// Strategy:
//  * Fold emb -> xcomb -> W_ih into per-vocab tables GXT_L/GXT_A (C,V,768):
//    gx = GXT_L[left] + GXT_A[above] + cb + r*PR + k*PC   (pure gathers).
//  * Fold target-embedding parts of green/blue heads into V x V tables.
//  * Wavefront recurrence: 1 workgroup per column (31 WGs x 256 thr = 8 waves),
//    cross-column sync via release/acquire progress counters.
//    Per GRU cell: two bf16 WMMA GEMMs (128x512x256 and 128x256x768),
//    fp32 accumulate, GEMM1 result staged in LDS for GEMM2 A-fragments.
//  * Heads: 961 WGs, 3 bf16 WMMA GEMMs each (N padded to 272), fused
//    bias/table adds, write (B,3,Nr,Nc,V) fp32 output.
// ---------------------------------------------------------------------------

#define NR 31
#define NC 31
#define BB 128
#define HH 256
#define EE 256
#define VV 257
#define CC 3
#define NPAD 272   // 257 padded up to 17 tiles of 16

typedef __attribute__((ext_vector_type(16))) __bf16 v16bf;
typedef __attribute__((ext_vector_type(8)))  float  v8f;

union FragU {
  v16bf v;
  uint4 q[2];
};

__device__ __forceinline__ v16bf frag_zero() {
  FragU f;
  f.q[0] = make_uint4(0u, 0u, 0u, 0u);
  f.q[1] = make_uint4(0u, 0u, 0u, 0u);
  return f.v;
}

// A fragment (16x32 bf16): lane row = lane&15, elems j: K = 8*(lane>>4) + (j&7) + 16*(j>>3).
// Two contiguous 16B chunks at +0 and +16 (in elements).
__device__ __forceinline__ v16bf load_a16(const unsigned short* p) {
  FragU f;
  f.q[0] = *(const uint4*)(p);
  f.q[1] = *(const uint4*)(p + 16);
  return f.v;
}

// B fragment (32x16 bf16): lane col = lane&15, elems j: K = 8*(lane>>4) + (j&7) + 16*(j>>3).
// Two contiguous 16B chunks at +0 and +16 along the weight row (row-major W[N][K]).
__device__ __forceinline__ v16bf load_b16(const unsigned short* p) {
  FragU f;
  f.q[0] = *(const uint4*)(p);
  f.q[1] = *(const uint4*)(p + 16);
  return f.v;
}

__device__ __forceinline__ v8f wmma_bf16(v16bf a, v16bf b, v8f c) {
  return __builtin_amdgcn_wmma_f32_16x16x32_bf16(false, a, false, b, (short)0, c,
                                                 false, false);
}

__device__ __forceinline__ v8f v8f_zero() {
  v8f z;
#pragma unroll
  for (int i = 0; i < 8; ++i) z[i] = 0.0f;
  return z;
}

__device__ __forceinline__ unsigned short f2bf(float x) {
  uint32_t u = __float_as_uint(x);
  u += 0x7FFFu + ((u >> 16) & 1u);
  return (unsigned short)(u >> 16);
}

__device__ __forceinline__ float sigmoidf_(float x) {
  return 1.0f / (1.0f + __expf(-x));
}

// ---------------------------------------------------------------------------
// Precompute kernels (tiny fp32 work; done once per launch)
// ---------------------------------------------------------------------------

__global__ void k_zero32(unsigned int* p, int n) {
  int i = blockIdx.x * blockDim.x + threadIdx.x;
  if (i < n) p[i] = 0u;
}

__global__ void k_cvt(const float* __restrict__ src, unsigned short* __restrict__ dst, int n) {
  int i = blockIdx.x * blockDim.x + threadIdx.x;
  if (i < n) dst[i] = f2bf(src[i]);
}

// dst: NPAD x 256 bf16, rows >= rows_src zero-filled.
__global__ void k_cvt_pad(const float* __restrict__ src, unsigned short* __restrict__ dst,
                          int rows_src, int ld_src) {
  int i = blockIdx.x * blockDim.x + threadIdx.x;
  if (i >= NPAD * HH) return;
  int n = i / HH, kk = i % HH;
  dst[i] = (n < rows_src) ? f2bf(src[(size_t)n * ld_src + kk]) : (unsigned short)0;
}

// T1_L[c][v][f] = sum_e xcomb_W[c][f][e]     * emb[c][v][e]
// T1_A[c][v][f] = sum_e xcomb_W[c][f][256+e] * emb[c][v][e]
__global__ void k_t1(const float* __restrict__ emb, const float* __restrict__ xcombW,
                     float* __restrict__ t1l, float* __restrict__ t1a) {
  int i = blockIdx.x * blockDim.x + threadIdx.x;
  if (i >= CC * VV * EE) return;
  int c = i / (VV * EE);
  int v = (i / EE) % VV;
  int f = i % EE;
  const float* e = emb + (size_t)(c * VV + v) * EE;
  const float* w = xcombW + (size_t)(c * EE + f) * (2 * EE);
  float sl = 0.f, sa = 0.f;
  for (int j = 0; j < EE; ++j) {
    sl += w[j] * e[j];
    sa += w[EE + j] * e[j];
  }
  t1l[i] = sl;
  t1a[i] = sa;
}

// GXT_L[c][v][g] = sum_f W_ih[c][g][f] * T1_L[c][v][f]   (same for A)
__global__ void k_gxt(const float* __restrict__ W_ih, const float* __restrict__ t1l,
                      const float* __restrict__ t1a, float* __restrict__ gxtl,
                      float* __restrict__ gxta) {
  int i = blockIdx.x * blockDim.x + threadIdx.x;
  if (i >= CC * VV * 768) return;
  int c = i / (VV * 768);
  int v = (i / 768) % VV;
  int g = i % 768;
  const float* w = W_ih + (size_t)(c * 768 + g) * 258;
  const float* l = t1l + (size_t)(c * VV + v) * EE;
  const float* a = t1a + (size_t)(c * VV + v) * EE;
  float sl = 0.f, sa = 0.f;
  for (int f = 0; f < EE; ++f) {
    sl += w[f] * l[f];
    sa += w[f] * a[f];
  }
  gxtl[i] = sl;
  gxta[i] = sa;
}

// cb[c][g] = b_ih[c][g] + sum_f W_ih[c][g][f] * xcomb_b[c][f];  PR/PC = pos columns.
__global__ void k_cbprc(const float* __restrict__ W_ih, const float* __restrict__ b_ih,
                        const float* __restrict__ xcomb_b, float* __restrict__ cb,
                        float* __restrict__ pr, float* __restrict__ pc) {
  int i = blockIdx.x * blockDim.x + threadIdx.x;
  if (i >= CC * 768) return;
  int c = i / 768;
  const float* w = W_ih + (size_t)i * 258;
  float s = b_ih[i];
  for (int f = 0; f < EE; ++f) s += w[f] * xcomb_b[c * EE + f];
  cb[i] = s;
  pr[i] = w[256];
  pc[i] = w[257];
}

// GT[v][u]  = sum_e emb[0][v][e] * green_W[u][256+e]
// BT0[v][u] = sum_e emb[0][v][e] * blue_W[u][256+e]
// BT1[v][u] = sum_e emb[1][v][e] * blue_W[u][512+e]
__global__ void k_tables(const float* __restrict__ emb, const float* __restrict__ green_W,
                         const float* __restrict__ blue_W, float* __restrict__ GT,
                         float* __restrict__ BT0, float* __restrict__ BT1) {
  int i = blockIdx.x * blockDim.x + threadIdx.x;
  if (i >= VV * VV) return;
  int v = i / VV, u = i % VV;
  const float* e0 = emb + (size_t)v * EE;
  const float* e1 = emb + (size_t)(VV + v) * EE;
  const float* gw = green_W + (size_t)u * 512 + 256;
  const float* bw = blue_W + (size_t)u * 768;
  float sg = 0.f, s0 = 0.f, s1 = 0.f;
  for (int e = 0; e < EE; ++e) {
    sg += e0[e] * gw[e];
    s0 += e0[e] * bw[256 + e];
    s1 += e1[e] * bw[512 + e];
  }
  GT[i] = sg;
  BT0[i] = s0;
  BT1[i] = s1;
}

// ---------------------------------------------------------------------------
// Wavefront GRU recurrence: one workgroup per column k.
// ---------------------------------------------------------------------------

__global__ __launch_bounds__(256) void rnn_kernel(
    const int* __restrict__ x,                  // (B,C,32,32)
    const unsigned short* __restrict__ hcombW,  // (C,256,512) bf16
    const float* __restrict__ hcomb_b,          // (C,256)
    const unsigned short* __restrict__ whhW,    // (C,768,256) bf16
    const float* __restrict__ b_hh,             // (C,768)
    const float* __restrict__ gxtL,             // (C,V,768)
    const float* __restrict__ gxtA,             // (C,V,768)
    const float* __restrict__ cb,               // (C,768)
    const float* __restrict__ pr, const float* __restrict__ pc,
    unsigned short* __restrict__ h_all,         // (NR,NC,C,128,256) bf16
    unsigned int* __restrict__ progress)        // (NC)
{
  __shared__ unsigned short h_stage[BB * HH];   // 64KB: GEMM1 output, GEMM2 A operand
  __shared__ int lidx[BB];
  __shared__ int aidx[BB];
  __shared__ float gxc[3 * HH];                 // cb + r*PR + k*PC

  const int k = blockIdx.x;
  const int tid = threadIdx.x;
  const int wave = tid >> 5;
  const int lane = tid & 31;
  const int ln = lane & 15;
  const int hi = lane >> 4;
  const int t0n = (2 * wave) * 16 + ln;         // this wave's N columns (tile 0/1)
  const int t1n = (2 * wave + 1) * 16 + ln;

  for (int r = 0; r < NR; ++r) {
    // wait for left neighbor column to finish row r
    if (tid == 0 && k > 0) {
      while (__hip_atomic_load(&progress[k - 1], __ATOMIC_ACQUIRE,
                               __HIP_MEMORY_SCOPE_AGENT) <= (unsigned)r) {
        __builtin_amdgcn_s_sleep(4);
      }
    }
    __syncthreads();

    for (int c = 0; c < CC; ++c) {
      // --- stage per-cell scalars -------------------------------------
      if (tid < BB) {
        int b = tid;
        lidx[b] = x[((b * CC + c) * 32 + (r + 1)) * 32 + k];
        aidx[b] = x[((b * CC + c) * 32 + r) * 32 + (k + 1)];
      }
      {
        float fr = (float)r, fk = (float)k;
        for (int g = tid; g < 3 * HH; g += 256)
          gxc[g] = cb[c * 768 + g] + fr * pr[c * 768 + g] + fk * pc[c * 768 + g];
      }
      __syncthreads();

      const unsigned short* leftB =
          (c > 0) ? (h_all + (((size_t)(r * NC + k) * CC + (c - 1)) << 15))
                  : ((k > 0) ? (h_all + (((size_t)(r * NC + (k - 1)) * CC + 2) << 15))
                             : (const unsigned short*)nullptr);
      const unsigned short* aboveB =
          (r > 0) ? (h_all + (((size_t)((r - 1) * NC + k) * CC + c) << 15))
                  : (const unsigned short*)nullptr;
      unsigned short* cellOut = h_all + (((size_t)(r * NC + k) * CC + c) << 15);

      // --- GEMM1: h = [h_left | h_above](128x512) @ hcombW[c]^T -------
      const unsigned short* hw = hcombW + (size_t)c * (HH * 2 * HH);
      v8f acc1[8][2];
#pragma unroll
      for (int m = 0; m < 8; ++m) {
        acc1[m][0] = v8f_zero();
        acc1[m][1] = v8f_zero();
      }
#pragma unroll 1
      for (int kk = 0; kk < 16; ++kk) {
        const unsigned short* ab = (kk < 8) ? leftB : aboveB;
        const int kl = (kk & 7) * 32 + hi * 8;   // K offset within 256-wide src
        const int kb = kk * 32 + hi * 8;         // K offset within 512-wide weight row
        v16bf b0 = load_b16(hw + (size_t)t0n * 512 + kb);
        v16bf b1 = load_b16(hw + (size_t)t1n * 512 + kb);
        if (kk + 1 < 16)
          __builtin_prefetch(hw + (size_t)t0n * 512 + (kk + 1) * 32, 0, 1);
        if (ab != nullptr) {
#pragma unroll
          for (int m = 0; m < 8; ++m) {
            v16bf a = load_a16(ab + (size_t)(m * 16 + ln) * HH + kl);
            acc1[m][0] = wmma_bf16(a, b0, acc1[m][0]);
            acc1[m][1] = wmma_bf16(a, b1, acc1[m][1]);
          }
        } else {
          v16bf a = frag_zero();
#pragma unroll
          for (int m = 0; m < 8; ++m) {
            acc1[m][0] = wmma_bf16(a, b0, acc1[m][0]);
            acc1[m][1] = wmma_bf16(a, b1, acc1[m][1]);
          }
        }
      }

      // add hcomb bias, stage h (bf16) to LDS for GEMM2 A operand
      {
        float bias0 = hcomb_b[c * HH + t0n];
        float bias1 = hcomb_b[c * HH + t1n];
#pragma unroll
        for (int m = 0; m < 8; ++m) {
#pragma unroll
          for (int i = 0; i < 8; ++i) {
            float h0v = acc1[m][0][i] + bias0;
            float h1v = acc1[m][1][i] + bias1;
            acc1[m][0][i] = h0v;
            acc1[m][1][i] = h1v;
            int row = m * 16 + i + 8 * hi;      // C-layout: M = i + 8*(lane>>4)
            h_stage[row * HH + t0n] = f2bf(h0v);
            h_stage[row * HH + t1n] = f2bf(h1v);
          }
        }
      }
      __syncthreads();

      // --- GEMM2: gh = h(128x256) @ W_hh[c]^T (256x768) + GRU update --
      const unsigned short* ww = whhW + (size_t)c * (768 * HH);
#pragma unroll 1
      for (int m = 0; m < 8; ++m) {
        v8f acc2[3][2];
#pragma unroll
        for (int g = 0; g < 3; ++g) {
          acc2[g][0] = v8f_zero();
          acc2[g][1] = v8f_zero();
        }
#pragma unroll 1
        for (int kk = 0; kk < 8; ++kk) {
          const unsigned short* rp = &h_stage[(m * 16 + ln) * HH + kk * 32 + hi * 8];
          v16bf a = load_a16(rp);               // ds_load_b128 x2
          const int kb = kk * 32 + hi * 8;
#pragma unroll
          for (int g = 0; g < 3; ++g) {
            v16bf bw0 = load_b16(ww + (size_t)(g * HH + t0n) * HH + kb);
            v16bf bw1 = load_b16(ww + (size_t)(g * HH + t1n) * HH + kb);
            acc2[g][0] = wmma_bf16(a, bw0, acc2[g][0]);
            acc2[g][1] = wmma_bf16(a, bw1, acc2[g][1]);
          }
        }
        // GRU elementwise for this M tile (columns owned by this wave)
#pragma unroll
        for (int t = 0; t < 2; ++t) {
          const int nH = (2 * wave + t) * 16 + ln;
          const float bh_r = b_hh[c * 768 + nH];
          const float bh_z = b_hh[c * 768 + nH + 256];
          const float bh_n = b_hh[c * 768 + nH + 512];
          const float gc_r = gxc[nH], gc_z = gxc[nH + 256], gc_n = gxc[nH + 512];
#pragma unroll
          for (int i = 0; i < 8; ++i) {
            const int brow = m * 16 + i + 8 * hi;
            const float* gl = gxtL + (size_t)(c * VV + lidx[brow]) * 768;
            const float* ga = gxtA + (size_t)(c * VV + aidx[brow]) * 768;
            float gxr = gl[nH] + ga[nH] + gc_r;
            float gxz = gl[nH + 256] + ga[nH + 256] + gc_z;
            float gxn = gl[nH + 512] + ga[nH + 512] + gc_n;
            float ghr = acc2[0][t][i] + bh_r;
            float ghz = acc2[1][t][i] + bh_z;
            float ghn = acc2[2][t][i] + bh_n;
            float rr = sigmoidf_(gxr + ghr);
            float zz = sigmoidf_(gxz + ghz);
            float nn = tanhf(gxn + rr * ghn);
            float hv = acc1[m][t][i];
            float hnew = (1.0f - zz) * nn + zz * hv;
            cellOut[(size_t)brow * HH + nH] = f2bf(hnew);
          }
        }
      }
      __threadfence();
      __syncthreads();
    }  // channel

    if (tid == 0) {
      __hip_atomic_store(&progress[k], (unsigned)(r + 1), __ATOMIC_RELEASE,
                         __HIP_MEMORY_SCOPE_AGENT);
    }
  }  // row
}

// ---------------------------------------------------------------------------
// Output heads: per (r,k), three 128x256x257 bf16 WMMA GEMMs + fused tables.
// ---------------------------------------------------------------------------

__global__ __launch_bounds__(256) void head_kernel(
    const unsigned short* __restrict__ h_all, const int* __restrict__ x,
    const unsigned short* __restrict__ redW,    // (NPAD,256) bf16, zero-padded
    const unsigned short* __restrict__ greenW,  // (NPAD,256) bf16 (first H cols)
    const unsigned short* __restrict__ blueW,   // (NPAD,256) bf16 (first H cols)
    const float* __restrict__ red_b, const float* __restrict__ green_b,
    const float* __restrict__ blue_b, const float* __restrict__ GT,
    const float* __restrict__ BT0, const float* __restrict__ BT1,
    float* __restrict__ out)                    // (B,3,NR,NC,V)
{
  __shared__ int t0s[BB];
  __shared__ int t1s[BB];
  const int blk = blockIdx.x;
  const int r = blk / NC, k = blk % NC;
  const int tid = threadIdx.x;
  const int wave = tid >> 5, lane = tid & 31, ln = lane & 15, hi = lane >> 4;

  if (tid < BB) {
    t0s[tid] = x[((tid * CC + 0) * 32 + (r + 1)) * 32 + (k + 1)];
    t1s[tid] = x[((tid * CC + 1) * 32 + (r + 1)) * 32 + (k + 1)];
  }
  __syncthreads();

#pragma unroll 1
  for (int head = 0; head < 3; ++head) {
    const unsigned short* A = h_all + (((size_t)(r * NC + k) * CC + head) << 15);
    const unsigned short* W = (head == 0) ? redW : (head == 1) ? greenW : blueW;
    const float* bias = (head == 0) ? red_b : (head == 1) ? green_b : blue_b;

    for (int t = wave; t < 17; t += 8) {
      const int n = t * 16 + ln;
#pragma unroll 1
      for (int m = 0; m < 8; ++m) {
        v8f acc = v8f_zero();
#pragma unroll 1
        for (int kk = 0; kk < 8; ++kk) {
          v16bf a = load_a16(A + (size_t)(m * 16 + ln) * HH + kk * 32 + hi * 8);
          v16bf b = load_b16(W + (size_t)n * HH + kk * 32 + hi * 8);
          acc = wmma_bf16(a, b, acc);
        }
        if (n < VV) {
          const float bn = bias[n];
#pragma unroll
          for (int i = 0; i < 8; ++i) {
            const int brow = m * 16 + i + 8 * hi;
            float v = acc[i] + bn;
            if (head == 1) v += GT[(size_t)t0s[brow] * VV + n];
            if (head == 2)
              v += BT0[(size_t)t0s[brow] * VV + n] + BT1[(size_t)t1s[brow] * VV + n];
            out[(((size_t)(brow * 3 + head) * NR + r) * NC + k) * VV + n] = v;
          }
        }
      }
    }
  }
}

// ---------------------------------------------------------------------------
// Host launcher
// ---------------------------------------------------------------------------

extern "C" void kernel_launch(void* const* d_in, const int* in_sizes, int n_in,
                              void* d_out, int out_size, void* d_ws, size_t ws_size,
                              hipStream_t stream) {
  (void)in_sizes; (void)n_in; (void)out_size; (void)ws_size;
  const int*   x        = (const int*)d_in[0];
  const float* emb      = (const float*)d_in[1];
  const float* xcomb_W  = (const float*)d_in[2];
  const float* xcomb_b  = (const float*)d_in[3];
  const float* hcomb_W  = (const float*)d_in[4];
  const float* hcomb_b  = (const float*)d_in[5];
  const float* W_ih     = (const float*)d_in[6];
  const float* W_hh     = (const float*)d_in[7];
  const float* b_ih     = (const float*)d_in[8];
  const float* b_hh     = (const float*)d_in[9];
  const float* red_W    = (const float*)d_in[10];
  const float* red_b    = (const float*)d_in[11];
  const float* green_W  = (const float*)d_in[12];
  const float* green_b  = (const float*)d_in[13];
  const float* blue_W   = (const float*)d_in[14];
  const float* blue_b   = (const float*)d_in[15];
  float* out = (float*)d_out;

  // ---- carve workspace (deterministic layout) ----
  size_t off = 0;
  char* base = (char*)d_ws;
  auto alloc = [&](size_t bytes) -> void* {
    void* p = base + off;
    off = (off + bytes + 255) & ~(size_t)255;
    return p;
  };
  unsigned int*   progress  = (unsigned int*)alloc(64 * sizeof(unsigned int));
  float*          t1l       = (float*)alloc((size_t)CC * VV * EE * 4);
  float*          t1a       = (float*)alloc((size_t)CC * VV * EE * 4);
  float*          gxtl      = (float*)alloc((size_t)CC * VV * 768 * 4);
  float*          gxta      = (float*)alloc((size_t)CC * VV * 768 * 4);
  float*          cb        = (float*)alloc((size_t)CC * 768 * 4);
  float*          pr        = (float*)alloc((size_t)CC * 768 * 4);
  float*          pc        = (float*)alloc((size_t)CC * 768 * 4);
  float*          GT        = (float*)alloc((size_t)VV * VV * 4);
  float*          BT0       = (float*)alloc((size_t)VV * VV * 4);
  float*          BT1       = (float*)alloc((size_t)VV * VV * 4);
  unsigned short* hcombW_bf = (unsigned short*)alloc((size_t)CC * HH * 2 * HH * 2);
  unsigned short* whhW_bf   = (unsigned short*)alloc((size_t)CC * 768 * HH * 2);
  unsigned short* redW_bf   = (unsigned short*)alloc((size_t)NPAD * HH * 2);
  unsigned short* greenW_bf = (unsigned short*)alloc((size_t)NPAD * HH * 2);
  unsigned short* blueW_bf  = (unsigned short*)alloc((size_t)NPAD * HH * 2);
  unsigned short* h_all     = (unsigned short*)alloc((size_t)NR * NC * CC * BB * HH * 2);

  auto cdiv = [](int a, int b) { return (a + b - 1) / b; };

  // progress flags must be re-zeroed every call (deterministic replay)
  k_zero32<<<1, 64, 0, stream>>>(progress, 64);

  // bf16 weight packing
  {
    int n1 = CC * HH * 2 * HH;  // 393216
    k_cvt<<<cdiv(n1, 256), 256, 0, stream>>>(hcomb_W, hcombW_bf, n1);
    int n2 = CC * 768 * HH;     // 589824
    k_cvt<<<cdiv(n2, 256), 256, 0, stream>>>(W_hh, whhW_bf, n2);
    int np = NPAD * HH;         // 69632
    k_cvt_pad<<<cdiv(np, 256), 256, 0, stream>>>(red_W, redW_bf, VV, HH);
    k_cvt_pad<<<cdiv(np, 256), 256, 0, stream>>>(green_W, greenW_bf, VV, HH + EE);
    k_cvt_pad<<<cdiv(np, 256), 256, 0, stream>>>(blue_W, blueW_bf, VV, HH + 2 * EE);
  }

  // fold embeddings through xcomb/W_ih and head weights into tables
  {
    int nt1 = CC * VV * EE;
    k_t1<<<cdiv(nt1, 256), 256, 0, stream>>>(emb, xcomb_W, t1l, t1a);
    int ngx = CC * VV * 768;
    k_gxt<<<cdiv(ngx, 256), 256, 0, stream>>>(W_ih, t1l, t1a, gxtl, gxta);
    int ncb = CC * 768;
    k_cbprc<<<cdiv(ncb, 256), 256, 0, stream>>>(W_ih, b_ih, xcomb_b, cb, pr, pc);
    int ntb = VV * VV;
    k_tables<<<cdiv(ntb, 256), 256, 0, stream>>>(emb, green_W, blue_W, GT, BT0, BT1);
  }

  // wavefront GRU recurrence: one workgroup per column
  rnn_kernel<<<NC, 256, 0, stream>>>(x, hcombW_bf, hcomb_b, whhW_bf, b_hh, gxtl, gxta,
                                     cb, pr, pc, h_all, progress);

  // output heads
  head_kernel<<<NR * NC, 256, 0, stream>>>(h_all, x, redW_bf, greenW_bf, blueW_bf,
                                           red_b, green_b, blue_b, GT, BT0, BT1, out);
}
